// RPNTargets_2705829396906
// MI455X (gfx1250) — matrix operand order repaired and checked by reading, not compiled, not run on hardware
//
#include <hip/hip_runtime.h>

// ---------------- problem constants (from reference) ----------------
#define HS_C 50
#define WS_C 64
#define AN_C 9
#define NANCH (HS_C * WS_C * AN_C)   // 28800 anchors per image
#define NGT 64
#define NIMG 16
#define POS_CAP_C 128u
#define BATCH_C 256u

typedef __attribute__((ext_vector_type(16))) _Float16 v16h;
typedef __attribute__((ext_vector_type(8)))  float    v8f;

#if defined(__gfx1250__) && __has_builtin(__builtin_amdgcn_global_load_async_to_lds_b32)
#define USE_ASYNC_G2L 1
#else
#define USE_ASYNC_G2L 0
#endif

typedef __attribute__((address_space(1))) int g_int;
typedef __attribute__((address_space(3))) int l_int;

// Deterministic per-image random key (replaces jax.random.uniform stream).
__device__ __forceinline__ unsigned keyhash(unsigned n, unsigned img, unsigned salt) {
    unsigned x = n * 0x9E3779B1u ^ (img * 0x85EBCA77u) ^ salt;
    x ^= x >> 16; x *= 0x7FEB352Du;
    x ^= x >> 15; x *= 0x846CA68Bu;
    x ^= x >> 16;
    return x;
}

// Analytic anchor box (normalized x1y1x2y2, zeroed when out of image).
__device__ __forceinline__ void anchor_box(unsigned n, float& x1, float& y1, float& x2, float& y2) {
    unsigned a    = n % AN_C;
    unsigned cell = n / AN_C;
    unsigned gx   = cell % WS_C;
    unsigned gy   = cell / WS_C;
    const float sc[3] = {128.0f, 256.0f, 512.0f};
    const float sr[3] = {0.70710678118654752f, 1.0f, 1.41421356237309515f};
    float w  = sc[a / 3] * sr[a % 3];
    float h  = sc[a / 3] / sr[a % 3];
    float cx = ((float)gx + 0.5f) * 16.0f;
    float cy = ((float)gy + 0.5f) * 16.0f;
    x1 = (cx - 0.5f * w) * (1.0f / 1024.0f);
    y1 = (cy - 0.5f * h) * (1.0f / 800.0f);
    x2 = (cx + 0.5f * w) * (1.0f / 1024.0f);
    y2 = (cy + 0.5f * h) * (1.0f / 800.0f);
    bool valid = (x1 > 0.0f) && (y1 > 0.0f) && (x2 < 1.0f) && (y2 < 1.0f);
    if (!valid) { x1 = 0.0f; y1 = 0.0f; x2 = 0.0f; y2 = 0.0f; }
}

// Tensor-core inclusive scan of a 256-bin histogram: D = L(16x16 lower-tri) x X,
// counts split into hi/lo bytes so every f16 input is integer-exact, f32 accum.
// Must be executed by one full wave32 (EXEC all ones).
__device__ __forceinline__ void wmma_scan256(const unsigned* hist, unsigned* incl, unsigned lane) {
    unsigned m  = lane & 15u;   // M row (A) / N column (B,D)
    unsigned hi = lane >> 4;

    // A = [L | 0] (16x32 f16). A-layout: lanes0-15 hold K0-15 (element e = K e),
    // lanes16-31 hold K16-31 (zero padding region).
    v16h A;
#pragma unroll
    for (int e = 0; e < 16; ++e) {
        int k = e + (hi ? 16 : 0);
        A[e] = (_Float16)((k < 16 && k <= (int)m) ? 1.0f : 0.0f);
    }
    // B (32x16 f16): lanes 0-15 carry K=0..15 for column N=lane; lanes 16-31 carry K=16..31 (zero pad).
    v16h Blo, Bhi;
#pragma unroll
    for (int e = 0; e < 16; ++e) {
        unsigned hval = hi ? 0u : hist[m * 16u + (unsigned)e];
        Blo[e] = (_Float16)(float)(hval & 255u);
        Bhi[e] = (_Float16)(float)(hval >> 8);
    }
    v8f zero = {};
    v8f Dlo = __builtin_amdgcn_wmma_f32_16x16x32_f16(false, A, false, Blo, (short)0, zero, false, false);
    v8f Dhi = __builtin_amdgcn_wmma_f32_16x16x32_f16(false, A, false, Bhi, (short)0, zero, false, false);

    float colscan[8];
#pragma unroll
    for (int i = 0; i < 8; ++i) colscan[i] = Dhi[i] * 256.0f + Dlo[i];

    // column totals live at M=15 -> VGPR7 of lanes 16..31 (column = lane-16)
    float mytot = colscan[7];
    float off = 0.0f;
#pragma unroll
    for (int c = 0; c < 15; ++c) {
        float t = __shfl(mytot, 16 + c, 32);
        if ((unsigned)c < m) off += t;
    }
#pragma unroll
    for (int i = 0; i < 8; ++i) {
        unsigned M = (unsigned)i + hi * 8u;
        unsigned b = m * 16u + M;        // bucket = column*16 + row
        incl[b] = (unsigned)(colscan[i] + off + 0.5f);
    }
}

// Exact descending top-k threshold via 4-level 256-ary radix select on 32-bit keys.
// Result: select candidate iff key > res.kv, plus res.rem more among key==kv in index order.
// res[0]=kv, res[1]=rem, res[2]=select_all, res[3]=total candidates.
__device__ void select_topk(unsigned img, unsigned n_target, unsigned salt, bool pos,
                            const unsigned char* s_info, const unsigned* s_forced,
                            unsigned* s_hist, unsigned* s_incl, unsigned* s_scal,
                            unsigned* res, unsigned tid, unsigned NT) {
    for (int level = 0; level < 4; ++level) {
        for (unsigned b = tid; b < 256u; b += NT) s_hist[b] = 0u;
        __syncthreads();

        bool skip = (level > 0) && (s_scal[2] != 0u);
        if (!skip) {
            const int shift = 24 - 8 * level;
            const unsigned pfx = (level > 0) ? s_scal[1] : 0u;
            for (unsigned n = tid; n < NANCH; n += NT) {
                unsigned ib = s_info[n];
                bool cand = pos ? (((ib & 0x40u) != 0u) || (((s_forced[n >> 5] >> (n & 31u)) & 1u) != 0u))
                                : ((ib & 0x80u) != 0u);
                if (!cand) continue;
                unsigned key = keyhash(n, img, salt);
                if (level > 0 && (key >> (unsigned)(shift + 8)) != pfx) continue;
                atomicAdd(&s_hist[(key >> (unsigned)shift) & 255u], 1u);
            }
        }
        __syncthreads();

        if (!skip && tid < 32u) wmma_scan256(s_hist, s_incl, tid);   // full wave0, EXEC all-ones
        __syncthreads();

        if (tid == 0u) {
            if (level == 0) {
                unsigned T = s_incl[255];
                s_scal[3] = T;
                if (T <= n_target) { s_scal[2] = 1u; s_scal[0] = 0u; s_scal[1] = 0u; }
                else               { s_scal[2] = 0u; s_scal[0] = n_target; s_scal[1] = 0u; }
            }
            if (s_scal[2] == 0u) {
                unsigned T_l  = s_incl[255];
                unsigned need = s_scal[0];
                int t = 0;
                for (int b = 255; b >= 0; --b) {
                    unsigned ib2 = s_incl[b];
                    unsigned hb  = ib2 - (b ? s_incl[b - 1] : 0u);
                    if (T_l - ib2 + hb >= need) { t = b; break; }
                }
                s_scal[0] = need - (T_l - s_incl[t]);
                s_scal[1] = (s_scal[1] << 8) | (unsigned)t;
            }
        }
        __syncthreads();
    }
    if (tid == 0u) { res[0] = s_scal[1]; res[1] = s_scal[0]; res[2] = s_scal[2]; res[3] = s_scal[3]; }
    __syncthreads();
}

__global__ void __launch_bounds__(1024)
rpn_targets_fused(const float* __restrict__ x, float* __restrict__ out) {
    __shared__ float              s_gt[NGT * 4];          // 1 KB
    __shared__ unsigned long long s_best[NGT];            // per-GT argmax (packed iou|~idx)
    __shared__ unsigned char      s_info[NANCH];          // [5:0]=matched idx, 0x40=pos-cand, 0x80=neg-cand
    __shared__ unsigned           s_forced[NANCH / 32];   // forced-positive bitmask
    __shared__ unsigned           s_hist[256];
    __shared__ unsigned           s_incl[256];
    __shared__ unsigned           s_scal[4];
    __shared__ unsigned           s_res[8];               // pos{kv,rem,all,total}, neg{...}

    const unsigned tid = threadIdx.x;
    const unsigned img = blockIdx.x;
    const unsigned NT  = blockDim.x;
    const unsigned SALT_P = 0x51E2D4C3u, SALT_N = 0xA7B3C9E1u;

    // ---- Phase 0: init + stage GT boxes (strided gather x[...,21:25]) into LDS ----
    for (unsigned i = tid; i < NGT; i += NT)        s_best[i]   = 0ull;
    for (unsigned i = tid; i < NANCH / 32; i += NT) s_forced[i] = 0u;
    if (tid < NGT * 4) {
        unsigned mm = tid >> 2, cc = tid & 3u;
        const float* src = x + ((size_t)(img * NGT + mm) * 25u + 21u + cc);
#if USE_ASYNC_G2L
        __builtin_amdgcn_global_load_async_to_lds_b32(
            (g_int*)(float*)src,
            (l_int*)&s_gt[tid], 0, 0);
#else
        s_gt[tid] = *src;
#endif
    }
#if USE_ASYNC_G2L
#if __has_builtin(__builtin_amdgcn_s_wait_asynccnt)
    __builtin_amdgcn_s_wait_asynccnt(0);
#else
    asm volatile("s_wait_asynccnt 0" ::: "memory");
#endif
#endif
    __syncthreads();

    // ---- Phase 1: IoU vs 64 GTs, per-anchor max/argmax, per-GT best (monotonic u64 max) ----
    for (unsigned n = tid; n < NANCH; n += NT) {
        float ax1, ay1, ax2, ay2;
        anchor_box(n, ax1, ay1, ax2, ay2);
        float area_a = (ax2 - ax1) * (ay2 - ay1);
        float best = -1.0f; unsigned bidx = 0u;
        for (unsigned j = 0; j < NGT; ++j) {
            float gx1 = s_gt[j * 4 + 0], gy1 = s_gt[j * 4 + 1];
            float gx2 = s_gt[j * 4 + 2], gy2 = s_gt[j * 4 + 3];
            float ix1 = fmaxf(ax1, gx1), iy1 = fmaxf(ay1, gy1);
            float ix2 = fminf(ax2, gx2), iy2 = fminf(ay2, gy2);
            float iw = fmaxf(ix2 - ix1, 0.0f), ih = fmaxf(iy2 - iy1, 0.0f);
            float inter = iw * ih;
            float area_g = (gx2 - gx1) * (gy2 - gy1);
            float iou = inter / (area_a + area_g - inter + 1e-7f);
            if (iou > best) { best = iou; bidx = j; }          // strict > keeps first index (jnp.argmax)
            if (iou > 0.0f) {
                unsigned long long pack =
                    ((unsigned long long)__float_as_uint(iou) << 32) |
                    (unsigned long long)(0xFFFFFFFFu - n);      // ties -> lowest anchor index wins
                if (pack > *(volatile unsigned long long*)&s_best[j])
                    atomicMax(&s_best[j], pack);
            }
        }
        unsigned info = bidx;
        if (best > 0.7f) info |= 0x40u;
        if (best < 0.3f) info |= 0x80u;
        s_info[n] = (unsigned char)info;
    }
    __syncthreads();

    // ---- Phase 2: forced positives (best anchor per GT with iou > 0) ----
    if (tid < NGT) {
        unsigned long long p = s_best[tid];
        float biou = __uint_as_float((unsigned)(p >> 32));
        if (biou > 0.0f) {
            unsigned n = 0xFFFFFFFFu - (unsigned)(p & 0xFFFFFFFFull);
            atomicOr(&s_forced[n >> 5], 1u << (n & 31u));
        }
    }
    __syncthreads();

    // ---- Phase 3/4: radix top-k selection (WMMA histogram scans inside) ----
    select_topk(img, POS_CAP_C, SALT_P, true, s_info, s_forced, s_hist, s_incl, s_scal, &s_res[0], tid, NT);
    unsigned Ppos  = s_res[3];
    unsigned n_pos = (Ppos < POS_CAP_C) ? Ppos : POS_CAP_C;
    unsigned n_neg = BATCH_C - n_pos;
    select_topk(img, n_neg, SALT_N, false, s_info, s_forced, s_hist, s_incl, s_scal, &s_res[4], tid, NT);

    const unsigned kvp = s_res[0], remp = s_res[1], allp = s_res[2];
    const unsigned kvn = s_res[4], remn = s_res[5], alln = s_res[6];

    // ---- Phase 5: mark + encode + write [B,HS,WS,A,5] ----
    for (unsigned n = tid; n < NANCH; n += NT) {
        unsigned ib = s_info[n];
        bool fb    = ((s_forced[n >> 5] >> (n & 31u)) & 1u) != 0u;
        bool candp = ((ib & 0x40u) != 0u) || fb;
        bool candn = (ib & 0x80u) != 0u;
        bool selp = false, seln = false;
        if (candp) {
            if (allp) selp = true;
            else {
                unsigned key = keyhash(n, img, SALT_P);
                if (key > kvp) selp = true;
                else if (key == kvp) {   // rare: rank equal-key candidates by index
                    unsigned r = 0;
                    for (unsigned m2 = 0; m2 < n; ++m2) {
                        unsigned ibm = s_info[m2];
                        bool cp = ((ibm & 0x40u) != 0u) || (((s_forced[m2 >> 5] >> (m2 & 31u)) & 1u) != 0u);
                        if (cp && keyhash(m2, img, SALT_P) == kvp) ++r;
                    }
                    selp = (r < remp);
                }
            }
        }
        if (candn) {
            if (alln) seln = true;
            else {
                unsigned key = keyhash(n, img, SALT_N);
                if (key > kvn) seln = true;
                else if (key == kvn) {
                    unsigned r = 0;
                    for (unsigned m2 = 0; m2 < n; ++m2) {
                        if (((s_info[m2] & 0x80u) != 0u) && keyhash(m2, img, SALT_N) == kvn) ++r;
                    }
                    seln = (r < remn);
                }
            }
        }

        float cls = selp ? 1.0f : -1.0f;
        if (seln) cls = 0.0f;   // reference applies sel_neg override after sel_pos
        float tx = 0.0f, ty = 0.0f, tw = 0.0f, th = 0.0f;
        if (selp) {
            float ax1, ay1, ax2, ay2;
            anchor_box(n, ax1, ay1, ax2, ay2);
            unsigned j = ib & 63u;
            float gx1 = s_gt[j * 4 + 0], gy1 = s_gt[j * 4 + 1];
            float gx2 = s_gt[j * 4 + 2], gy2 = s_gt[j * 4 + 3];
            float w_gt = gx2 - gx1, h_gt = gy2 - gy1;
            float xc_gt = 0.5f * (gx1 + gx2), yc_gt = 0.5f * (gy1 + gy2);
            float w_r = ax2 - ax1, h_r = ay2 - ay1;
            float xc_r = 0.5f * (ax1 + ax2), yc_r = 0.5f * (ay1 + ay2);
            float w_rs = (w_r > 0.0f) ? w_r : 1.0f;
            float h_rs = (h_r > 0.0f) ? h_r : 1.0f;
            tx = (w_gt > 0.0f) ? (xc_gt - xc_r) / w_rs : 0.0f;
            ty = (h_gt > 0.0f) ? (yc_gt - yc_r) / h_rs : 0.0f;
            tw = (w_gt > 0.0f) ? logf(w_gt / w_rs) : 0.0f;
            th = (h_gt > 0.0f) ? logf(h_gt / h_rs) : 0.0f;
        }
        size_t base = ((size_t)img * NANCH + n) * 5u;
        out[base + 0] = cls;
        out[base + 1] = tx;
        out[base + 2] = ty;
        out[base + 3] = tw;
        out[base + 4] = th;
    }
}

extern "C" void kernel_launch(void* const* d_in, const int* in_sizes, int n_in,
                              void* d_out, int out_size, void* d_ws, size_t ws_size,
                              hipStream_t stream) {
    (void)in_sizes; (void)n_in; (void)out_size; (void)d_ws; (void)ws_size;
    const float* x = (const float*)d_in[0];
    float* out = (float*)d_out;
    rpn_targets_fused<<<NIMG, 1024, 0, stream>>>(x, out);
}
